// GAT_8701603741743
// MI455X (gfx1250) — matrix operand order repaired and checked by reading, not compiled, not run on hardware
//
#include <hip/hip_runtime.h>
#include <hip/hip_bf16.h>
#include <math.h>

// ---- problem constants (from reference) ----
#define GN   40000        // nodes
#define GE   640000       // edges (before self loops)
#define GETOT (GE + GN)   // edges incl. self loops
#define GF   128          // features per head (layers 0/1)
#define GH   2            // heads (layers 0/1)
#define GFH  (GF*GH)      // 256
#define GC   40           // output classes
#define GCPAD 48          // GC padded to tile multiple
#define NEG_SLOPE 0.2f

typedef __attribute__((ext_vector_type(16))) __bf16 v16bf;
typedef __attribute__((ext_vector_type(8)))  float  v8f;

union BOp { v16bf v; uint4 q[2]; };   // 32 bytes: one WMMA 16-bit operand per lane

__device__ __forceinline__ unsigned short f2bf_rne(float f) {
    unsigned u = __float_as_uint(f);
    unsigned r = u + 0x7FFFu + ((u >> 16) & 1u);   // round-to-nearest-even
    return (unsigned short)(r >> 16);
}
__device__ __forceinline__ float bf2f(unsigned short h) {
    return __uint_as_float(((unsigned)h) << 16);
}

// ---------------------------------------------------------------------------
// Prepack X[M,K] (row-major f32) -> bf16 hi/lo planes, same row-major layout.
// ---------------------------------------------------------------------------
__global__ void gat_pack_x(const float* __restrict__ X, unsigned short* __restrict__ Xh,
                           unsigned short* __restrict__ Xl, long long total) {
    long long t = (long long)blockIdx.x * blockDim.x + threadIdx.x;
    if (t >= total) return;
    float f = X[t];
    unsigned short h = f2bf_rne(f);
    Xh[t] = h;
    Xl[t] = f2bf_rne(f - bf2f(h));
}

// ---------------------------------------------------------------------------
// Prepack W[K,Nc] (row-major f32) -> transposed, zero-padded bf16 hi/lo planes
// WT[NcPad][K]: one GEMM B-column becomes one contiguous row.
// ---------------------------------------------------------------------------
__global__ void gat_pack_w(const float* __restrict__ W, unsigned short* __restrict__ WTh,
                           unsigned short* __restrict__ WTl, int K, int Nc, int NcPad) {
    int t = blockIdx.x * blockDim.x + threadIdx.x;
    if (t >= NcPad * K) return;
    int n = t / K, k = t - n * K;
    float f = (n < Nc) ? W[(size_t)k * Nc + n] : 0.0f;
    unsigned short h = f2bf_rne(f);
    WTh[t] = h;
    WTl[t] = f2bf_rne(f - bf2f(h));
}

// ---------------------------------------------------------------------------
// GEMM: Hout[M,Nc] = X[M,K] @ W[K,Nc] using prepacked bf16x3 operands.
// Template G (N-tiles per wave) and KC (reduction depth) so the hot loop has
// zero scalar branches: 4 b128 A-loads + G*(4 b128 B-loads) + 3G WMMAs / step.
// ---------------------------------------------------------------------------
template <int G, int KC>
__global__ void gat_gemm_wmma(const unsigned short* __restrict__ Xh,
                              const unsigned short* __restrict__ Xl,
                              const unsigned short* __restrict__ WTh,
                              const unsigned short* __restrict__ WTl,
                              float* __restrict__ Hout,
                              int M, int Nc, int numGroups) {
    int wave = blockIdx.x * (blockDim.x >> 5) + (threadIdx.x >> 5);
    int lane = threadIdx.x & 31;
    int tM = M >> 4;
    if (wave >= tM * numGroups) return;     // uniform whole-wave exit: EXEC stays all-1s
    int g  = wave % numGroups;
    int tm = wave / numGroups;
    int mbase = tm << 4;
    int nbase = g * G * 16;
    int l16 = lane & 15;
    bool hiHalf = lane >= 16;

    v8f c[G];
    #pragma unroll
    for (int t = 0; t < G; ++t) c[t] = (v8f){0.f,0.f,0.f,0.f,0.f,0.f,0.f,0.f};

    int mrow = mbase + l16;
    // A per-lane layout (ISA 16-bit A): us[0..7]=K off..off+7, us[8..15]=K off+16..off+23
    const unsigned short* arh = Xh + (size_t)mrow * KC + (hiHalf ? 8 : 0);
    const unsigned short* arl = Xl + (size_t)mrow * KC + (hiHalf ? 8 : 0);
    // B per-lane layout: us[0..7]=K off..off+7, us[8..15]=K off+16..off+23 of column nbase+t*16+l16
    size_t boff = (size_t)(nbase + l16) * KC + (hiHalf ? 8 : 0);
    const unsigned short* brh = WTh + boff;
    const unsigned short* brl = WTl + boff;
    const size_t tstride = (size_t)16 * KC;        // advance one N-tile

    for (int kb = 0; kb < KC; kb += 32) {
        __builtin_prefetch(arh + kb + 32, 0, 3);   // global_prefetch_b8 into near cache
        BOp ah, al;
        ah.q[0] = *(const uint4*)(arh + kb);
        ah.q[1] = *(const uint4*)(arh + kb + 16);
        al.q[0] = *(const uint4*)(arl + kb);
        al.q[1] = *(const uint4*)(arl + kb + 16);
        #pragma unroll
        for (int t = 0; t < G; ++t) {
            BOp bh, bl;
            const unsigned short* ph = brh + t * tstride + kb;
            const unsigned short* pl = brl + t * tstride + kb;
            bh.q[0] = *(const uint4*)(ph);
            bh.q[1] = *(const uint4*)(ph + 16);
            bl.q[0] = *(const uint4*)(pl);
            bl.q[1] = *(const uint4*)(pl + 16);
            c[t] = __builtin_amdgcn_wmma_f32_16x16x32_bf16(false, ah.v, false, bh.v, (short)0, c[t], false, false);
            c[t] = __builtin_amdgcn_wmma_f32_16x16x32_bf16(false, ah.v, false, bl.v, (short)0, c[t], false, false);
            c[t] = __builtin_amdgcn_wmma_f32_16x16x32_bf16(false, al.v, false, bh.v, (short)0, c[t], false, false);
        }
    }
    #pragma unroll
    for (int t = 0; t < G; ++t) {
        int ncol = nbase + t * 16 + l16;
        if (ncol >= Nc) continue;                    // ragged last tile (Nc=40)
        #pragma unroll
        for (int r = 0; r < 8; ++r) {                // C layout: vgpr r -> M=r / M=8+r
            int row = mbase + (hiHalf ? 8 + r : r);
            Hout[(size_t)row * Nc + ncol] = c[t][r];
        }
    }
}

// ---------------------------------------------------------------------------
// Per-node attention scores, wave per node: coalesced loads + shfl reduction.
// s_src[n,h] = <h[n,h,:], a_src[h,:]>, same for dst.
// ---------------------------------------------------------------------------
__global__ void gat_scores(const float* __restrict__ Hf, const float* __restrict__ a_src,
                           const float* __restrict__ a_dst, float* __restrict__ s_src,
                           float* __restrict__ s_dst, int n_nodes, int Hh, int F_) {
    int n = blockIdx.x * (blockDim.x >> 5) + (threadIdx.x >> 5);
    if (n >= n_nodes) return;
    int lane = threadIdx.x & 31;
    for (int h = 0; h < Hh; ++h) {
        const float* hp = Hf + (size_t)n * Hh * F_ + (size_t)h * F_;
        const float* as = a_src + h * F_;
        const float* ad = a_dst + h * F_;
        float ss = 0.f, sd = 0.f;
        for (int f = lane; f < F_; f += 32) {
            float v = hp[f];
            ss += v * as[f];
            sd += v * ad[f];
        }
        #pragma unroll
        for (int off = 16; off > 0; off >>= 1) {
            ss += __shfl_xor(ss, off, 32);
            sd += __shfl_xor(sd, off, 32);
        }
        if (lane == 0) {
            s_src[n * Hh + h] = ss;
            s_dst[n * Hh + h] = sd;
        }
    }
}

__device__ __forceinline__ void edge_sd(const int* __restrict__ ei, int e, int& s, int& d) {
    if (e < GE) { s = ei[e]; d = ei[GE + e]; }
    else        { s = d = e - GE; }          // implicit self loops
}

// pass 1: segment max of leaky-relu'd logits (ordered-uint atomicMax; m_u zeroed)
__global__ void gat_edge_max(const int* __restrict__ ei, const float* __restrict__ s_src,
                             const float* __restrict__ s_dst, unsigned* __restrict__ m_u, int Hh) {
    int t = blockIdx.x * blockDim.x + threadIdx.x;
    if (t >= GETOT * Hh) return;
    int e = t / Hh, h = t - e * Hh;
    int s, d; edge_sd(ei, e, s, d);
    float x = s_src[s * Hh + h] + s_dst[d * Hh + h];
    x = x > 0.f ? x : NEG_SLOPE * x;
    unsigned u = __float_as_uint(x);
    unsigned key = (u & 0x80000000u) ? ~u : (u | 0x80000000u);
    atomicMax(&m_u[d * Hh + h], key);
}

// pass 2: ex = exp(e - max); store per-edge numerator; segment-sum denominator
__global__ void gat_edge_exp(const int* __restrict__ ei, const float* __restrict__ s_src,
                             const float* __restrict__ s_dst, const unsigned* __restrict__ m_u,
                             float* __restrict__ exbuf, float* __restrict__ denom, int Hh) {
    int t = blockIdx.x * blockDim.x + threadIdx.x;
    if (t >= GETOT * Hh) return;
    int e = t / Hh, h = t - e * Hh;
    int s, d; edge_sd(ei, e, s, d);
    float x = s_src[s * Hh + h] + s_dst[d * Hh + h];
    x = x > 0.f ? x : NEG_SLOPE * x;
    unsigned k = m_u[d * Hh + h];
    float mf = (k & 0x80000000u) ? __uint_as_float(k ^ 0x80000000u) : __uint_as_float(~k);
    float ex = expf(x - mf);
    exbuf[t] = ex;
    atomicAdd(&denom[d * Hh + h], ex);
}

// pass 3: heavy scatter out[dst] += alpha * h[src]; one wave per edge (L2-resident)
__global__ void gat_edge_scatter(const int* __restrict__ ei, const float* __restrict__ Hf,
                                 const float* __restrict__ exbuf, const float* __restrict__ denom,
                                 float* __restrict__ Obuf, int Hh, int F_) {
    int e = blockIdx.x * (blockDim.x >> 5) + (threadIdx.x >> 5);
    if (e >= GETOT) return;
    int lane = threadIdx.x & 31;
    int FH = Hh * F_;
    int s, d; edge_sd(ei, e, s, d);
    const float* hsrc = Hf + (size_t)s * FH;
    float* odst = Obuf + (size_t)d * FH;
    for (int i = lane; i < FH; i += 32) {
        int h = i / F_;
        float alpha = exbuf[(size_t)e * Hh + h] / (denom[d * Hh + h] + 1e-16f);
        atomicAdd(&odst[i], hsrc[i] * alpha);
    }
}

__global__ void gat_bias_elu(float* __restrict__ A, const float* __restrict__ b, int total, int FH) {
    int t = blockIdx.x * blockDim.x + threadIdx.x;
    if (t >= total) return;
    float v = A[t] + b[t % FH];
    A[t] = v > 0.f ? v : (expf(v) - 1.0f);
}

__global__ void gat_bias_out(const float* __restrict__ A, const float* __restrict__ b,
                             float* __restrict__ out, int total, int Cc) {
    int t = blockIdx.x * blockDim.x + threadIdx.x;
    if (t >= total) return;
    out[t] = A[t] + b[t % Cc];
}

__global__ void gat_zero(float* __restrict__ p, long long n) {
    long long t = (long long)blockIdx.x * blockDim.x + threadIdx.x;
    long long stride = (long long)gridDim.x * blockDim.x;
    for (; t < n; t += stride) p[t] = 0.0f;
}

// ---------------------------------------------------------------------------
extern "C" void kernel_launch(void* const* d_in, const int* in_sizes, int n_in,
                              void* d_out, int out_size, void* d_ws, size_t ws_size,
                              hipStream_t stream) {
    const float* x      = (const float*)d_in[0];
    const int*   ei     = (const int*)  d_in[1];
    const float* W0     = (const float*)d_in[2];
    const float* a_src0 = (const float*)d_in[3];
    const float* a_dst0 = (const float*)d_in[4];
    const float* b0     = (const float*)d_in[5];
    const float* W1     = (const float*)d_in[6];
    const float* a_src1 = (const float*)d_in[7];
    const float* a_dst1 = (const float*)d_in[8];
    const float* b1     = (const float*)d_in[9];
    const float* Wo     = (const float*)d_in[10];
    const float* a_srco = (const float*)d_in[11];
    const float* a_dsto = (const float*)d_in[12];
    const float* bo     = (const float*)d_in[13];
    float* out = (float*)d_out;

    // workspace layout
    float*          Abuf  = (float*)d_ws;                     // N*FH activations / accumulator
    float*          Hbuf  = Abuf  + (size_t)GN * GFH;         // N*FH transformed features
    float*          s_src = Hbuf  + (size_t)GN * GFH;         // N*H
    float*          s_dst = s_src + (size_t)GN * GH;          // N*H
    unsigned*       m_u   = (unsigned*)(s_dst + (size_t)GN * GH);  // N*H
    float*          denom = (float*)m_u + (size_t)GN * GH;    // N*H
    float*          exbuf = denom + (size_t)GN * GH;          // Etot*H
    unsigned short* XPh   = (unsigned short*)(exbuf + (size_t)GETOT * GH); // N*FH bf16
    unsigned short* XPl   = XPh + (size_t)GN * GFH;
    unsigned short* WTh   = XPl + (size_t)GN * GFH;           // NcPad*K bf16 (max 256*256)
    unsigned short* WTl   = WTh + (size_t)GFH * GFH;

    const int BLK = 256;
    const int wavesPerBlk = BLK / 32;
    auto cdiv = [](long long a, long long b) { return (int)((a + b - 1) / b); };

    int edgeBlocksH  = cdiv((long long)GETOT * GH, BLK);
    int edgeBlocks1  = cdiv((long long)GETOT, BLK);
    int scatBlocks   = cdiv((long long)GETOT, wavesPerBlk);
    int nodeFHBlocks = cdiv((long long)GN * GFH, BLK);
    int nodeCBlocks  = cdiv((long long)GN * GC, BLK);
    int scoreBlocks  = cdiv((long long)GN, wavesPerBlk);
    int zeroBlocks   = 2048;
    // layers 0/1: NcPad=256 -> 16 N-tiles, G=4 -> 4 groups; out: NcPad=48 -> G=3, 1 group
    int gemmBlocks01 = cdiv((long long)(GN / 16) * 4, wavesPerBlk);
    int gemmBlocksO  = cdiv((long long)(GN / 16) * 1, wavesPerBlk);

    // ===================== layer 0 (in: x [N,128]) =====================
    gat_pack_x<<<cdiv((long long)GN * GF, BLK), BLK, 0, stream>>>(x, XPh, XPl, (long long)GN * GF);
    gat_pack_w<<<cdiv((long long)GFH * GF, BLK), BLK, 0, stream>>>(W0, WTh, WTl, GF, GFH, GFH);
    gat_gemm_wmma<4, GF><<<gemmBlocks01, BLK, 0, stream>>>(XPh, XPl, WTh, WTl, Hbuf, GN, GFH, 4);
    gat_scores<<<scoreBlocks, BLK, 0, stream>>>(Hbuf, a_src0, a_dst0, s_src, s_dst, GN, GH, GF);
    gat_zero<<<zeroBlocks, BLK, 0, stream>>>((float*)m_u, (long long)GN * GH * 2);  // m_u + denom
    gat_zero<<<zeroBlocks, BLK, 0, stream>>>(Abuf, (long long)GN * GFH);
    gat_edge_max<<<edgeBlocksH, BLK, 0, stream>>>(ei, s_src, s_dst, m_u, GH);
    gat_edge_exp<<<edgeBlocksH, BLK, 0, stream>>>(ei, s_src, s_dst, m_u, exbuf, denom, GH);
    gat_edge_scatter<<<scatBlocks, BLK, 0, stream>>>(ei, Hbuf, exbuf, denom, Abuf, GH, GF);
    gat_bias_elu<<<nodeFHBlocks, BLK, 0, stream>>>(Abuf, b0, GN * GFH, GFH);

    // ===================== layer 1 (in: Abuf [N,256]) ===================
    gat_pack_x<<<cdiv((long long)GN * GFH, BLK), BLK, 0, stream>>>(Abuf, XPh, XPl, (long long)GN * GFH);
    gat_pack_w<<<cdiv((long long)GFH * GFH, BLK), BLK, 0, stream>>>(W1, WTh, WTl, GFH, GFH, GFH);
    gat_gemm_wmma<4, GFH><<<gemmBlocks01, BLK, 0, stream>>>(XPh, XPl, WTh, WTl, Hbuf, GN, GFH, 4);
    gat_scores<<<scoreBlocks, BLK, 0, stream>>>(Hbuf, a_src1, a_dst1, s_src, s_dst, GN, GH, GF);
    gat_zero<<<zeroBlocks, BLK, 0, stream>>>((float*)m_u, (long long)GN * GH * 2);
    gat_zero<<<zeroBlocks, BLK, 0, stream>>>(Abuf, (long long)GN * GFH);   // GEMM consumed Abuf via XP*
    gat_edge_max<<<edgeBlocksH, BLK, 0, stream>>>(ei, s_src, s_dst, m_u, GH);
    gat_edge_exp<<<edgeBlocksH, BLK, 0, stream>>>(ei, s_src, s_dst, m_u, exbuf, denom, GH);
    gat_edge_scatter<<<scatBlocks, BLK, 0, stream>>>(ei, Hbuf, exbuf, denom, Abuf, GH, GF);
    gat_bias_elu<<<nodeFHBlocks, BLK, 0, stream>>>(Abuf, b1, GN * GFH, GFH);

    // ===================== output layer (heads=1, F=C=40) ===============
    gat_pack_x<<<cdiv((long long)GN * GFH, BLK), BLK, 0, stream>>>(Abuf, XPh, XPl, (long long)GN * GFH);
    gat_pack_w<<<cdiv((long long)GCPAD * GFH, BLK), BLK, 0, stream>>>(Wo, WTh, WTl, GFH, GC, GCPAD);
    gat_gemm_wmma<3, GFH><<<gemmBlocksO, BLK, 0, stream>>>(XPh, XPl, WTh, WTl, Hbuf, GN, GC, 1);
    gat_scores<<<scoreBlocks, BLK, 0, stream>>>(Hbuf, a_srco, a_dsto, s_src, s_dst, GN, 1, GC);
    gat_zero<<<zeroBlocks, BLK, 0, stream>>>((float*)m_u, (long long)GN * GH * 2);
    gat_zero<<<zeroBlocks, BLK, 0, stream>>>(Abuf, (long long)GN * GC);
    gat_edge_max<<<edgeBlocks1, BLK, 0, stream>>>(ei, s_src, s_dst, m_u, 1);
    gat_edge_exp<<<edgeBlocks1, BLK, 0, stream>>>(ei, s_src, s_dst, m_u, exbuf, denom, 1);
    gat_edge_scatter<<<scatBlocks, BLK, 0, stream>>>(ei, Hbuf, exbuf, denom, Abuf, 1, GC);
    gat_bias_out<<<nodeCBlocks, BLK, 0, stream>>>(Abuf, bo, out, GN * GC, GC);
}
